// FasterRCNNMobileNet_46634754900479
// MI455X (gfx1250) — matrix operand order repaired and checked by reading, not compiled, not run on hardware
//
#include <hip/hip_runtime.h>
#include <math.h>

// ---------------- problem constants (from reference) ----------------
#define ROI_N   1000
#define NPAD    1024          // padded M, 16 * 64  (divides 64-row wave blocks)
#define C_CH    256
#define FB      2
#define FH      100
#define FW      136
#define D_FEAT  12544         // 256 * 49, multiple of 32
#define HID     1024
#define IMGW    1088.0f
#define IMGH    800.0f

// GEMM register blocking: each wave computes (16*MT) x (16*NT) outputs
#define MT 4
#define NT 2

typedef __attribute__((ext_vector_type(16))) __bf16 v16bf;
typedef __attribute__((ext_vector_type(8)))  float  v8f;

// -------- bf16 helpers (bit-exact, no reliance on __bf16 codegen) --------
__device__ __forceinline__ unsigned short f32_to_bf16_rne(float x) {
  unsigned int u = __float_as_uint(x);
  unsigned int r = u + 0x7FFFu + ((u >> 16) & 1u);
  return (unsigned short)(r >> 16);
}
__device__ __forceinline__ float bf16_to_f32(unsigned short h) {
  return __uint_as_float(((unsigned int)h) << 16);
}
__device__ __forceinline__ void split_bf16(float v, unsigned short& hi, unsigned short& lo) {
  hi = f32_to_bf16_rne(v);
  lo = f32_to_bf16_rne(v - bf16_to_f32(hi));
}

// ================= 1. transpose + hi/lo split of weights =================
// in:  W [K, N] f32 (row-major, N contiguous)   out: WT_hi/lo [N, K] bf16
__global__ void transpose_split_kernel(const float* __restrict__ w,
                                       unsigned short* __restrict__ wt_hi,
                                       unsigned short* __restrict__ wt_lo,
                                       int K, int N) {
  __shared__ float tile[32][33];
  int k0 = blockIdx.x * 32;
  int n0 = blockIdx.y * 32;
  int tx = threadIdx.x, ty = threadIdx.y;
#pragma unroll
  for (int j = 0; j < 32; j += 8)
    tile[ty + j][tx] = w[(size_t)(k0 + ty + j) * N + (n0 + tx)];
  __syncthreads();
#pragma unroll
  for (int j = 0; j < 32; j += 8) {
    float v = tile[tx][ty + j];
    unsigned short hi, lo;
    split_bf16(v, hi, lo);
    size_t o = (size_t)(n0 + ty + j) * K + (k0 + tx);
    wt_hi[o] = hi;
    wt_lo[o] = lo;
  }
}

// ======================= 2. ROI-align (7x7, SR=2) ========================
struct SampleLDS { int o00, o01, o10, o11; float w00, w01, w10, w11; };

__global__ void roi_align_kernel(const float* __restrict__ feat,
                                 const float* __restrict__ boxes,
                                 const int*   __restrict__ roib,
                                 unsigned short* __restrict__ out_hi,
                                 unsigned short* __restrict__ out_lo) {
  __shared__ SampleLDS smp[196];
  int n = blockIdx.x;
  int c = threadIdx.x;           // one thread per channel (256)
  if (n >= ROI_N) {              // zero the pad rows so GEMM sees clean A
#pragma unroll 1
    for (int bin = 0; bin < 49; ++bin) {
      size_t o = (size_t)n * D_FEAT + c * 49 + bin;
      out_hi[o] = 0; out_lo[o] = 0;
    }
    return;
  }
  float x1 = boxes[n * 4 + 0] * 0.125f, y1 = boxes[n * 4 + 1] * 0.125f;
  float x2 = boxes[n * 4 + 2] * 0.125f, y2 = boxes[n * 4 + 3] * 0.125f;
  float rw = fmaxf(x2 - x1, 1.0f), rh = fmaxf(y2 - y1, 1.0f);
  float bw = rw * (1.0f / 7.0f), bh = rh * (1.0f / 7.0f);
  if (c < 196) {                 // precompute the 14x14 sample grid once
    int py = c / 14, px = c % 14;
    float offy = (float)(py >> 1) + ((float)(py & 1) + 0.5f) * 0.5f;
    float offx = (float)(px >> 1) + ((float)(px & 1) + 0.5f) * 0.5f;
    float y = fminf(fmaxf(y1 + offy * bh, 0.0f), (float)(FH - 1));
    float x = fminf(fmaxf(x1 + offx * bw, 0.0f), (float)(FW - 1));
    float y0f = floorf(y), x0f = floorf(x);
    int y0 = (int)y0f, x0 = (int)x0f;
    int y1i = min(y0 + 1, FH - 1), x1i = min(x0 + 1, FW - 1);
    float ly = y - y0f, lx = x - x0f;
    SampleLDS s;
    s.o00 = y0 * FW + x0;  s.o01 = y0 * FW + x1i;
    s.o10 = y1i * FW + x0; s.o11 = y1i * FW + x1i;
    s.w00 = (1.f - ly) * (1.f - lx); s.w01 = (1.f - ly) * lx;
    s.w10 = ly * (1.f - lx);         s.w11 = ly * lx;
    smp[c] = s;
  }
  __syncthreads();
  int b = roib[n];
  const float* plane = feat + ((size_t)b * C_CH + c) * (FH * FW);
#pragma unroll 1
  for (int oy = 0; oy < 7; ++oy)
#pragma unroll 1
    for (int ox = 0; ox < 7; ++ox) {
      float acc = 0.f;
#pragma unroll
      for (int sy = 0; sy < 2; ++sy)
#pragma unroll
        for (int sx = 0; sx < 2; ++sx) {
          SampleLDS s = smp[(oy * 2 + sy) * 14 + (ox * 2 + sx)];
          acc += s.w00 * plane[s.o00] + s.w01 * plane[s.o01]
               + s.w10 * plane[s.o10] + s.w11 * plane[s.o11];
        }
      unsigned short hi, lo;
      split_bf16(acc * 0.25f, hi, lo);
      size_t o = (size_t)n * D_FEAT + c * 49 + oy * 7 + ox;
      out_hi[o] = hi; out_lo[o] = lo;
    }
}

// ============ 3. split-bf16 WMMA GEMM, register-blocked (+bias+ReLU) ============
// C[M,NCOL] = relu(A[M,K] * W[K,NCOL] + bias), W given pre-transposed [NCOL,K].
// One wave computes a (16*MT) x (16*NT) block.  K is a compile-time constant so
// every fragment address is (one of 4 base pointers) + constant IOFFSET
// (mt*16*K*2 <= 1.2MB, inside the signed-24-bit global_load offset): no pointer
// arrays, no per-load address VALU -> no spills, no WMMA->VALU hazard NOPs.
// Split trick: A ~= Ahi+Alo, W ~= Whi+Wlo; C ~= Ahi*Whi + Ahi*Wlo + Alo*Whi.
union Frag16 { v16bf v; uint4 q[2]; };

template <int MODE, int KC>   // MODE 0: store split-bf16; 1: store f32
__global__ void gemm_wmma_kernel(const unsigned short* __restrict__ a_hi,
                                 const unsigned short* __restrict__ a_lo,
                                 const unsigned short* __restrict__ wt_hi,
                                 const unsigned short* __restrict__ wt_lo,
                                 const float* __restrict__ bias,
                                 unsigned short* __restrict__ out_hi,
                                 unsigned short* __restrict__ out_lo,
                                 float* __restrict__ out_f, int NCOL) {
  int m0 = blockIdx.x * (16 * MT);
  int n0 = blockIdx.y * (16 * NT);
  int lane = threadIdx.x;
  int lm = lane & 15;       // row (A) / col (B) within a 16x16 tile
  int lh = lane >> 4;       // half-wave selector (K interleave per ISA layout)

  // Four base pointers; all other addressing is compile-time instruction offsets.
  const unsigned short* pAH = a_hi + (size_t)(m0 + lm) * KC + 8 * lh;
  const unsigned short* pAL = a_lo + (size_t)(m0 + lm) * KC + 8 * lh;
  const unsigned short* pBH = wt_hi + (size_t)(n0 + lm) * KC + 8 * lh;
  const unsigned short* pBL = wt_lo + (size_t)(n0 + lm) * KC + 8 * lh;

  v8f acc[MT][NT];
#pragma unroll
  for (int mt = 0; mt < MT; ++mt)
#pragma unroll
    for (int nt = 0; nt < NT; ++nt) acc[mt][nt] = (v8f){};

#pragma unroll 1
  for (int k0 = 0; k0 < KC; k0 += 32) {
    Frag16 ah[MT], al[MT], bh[NT], bl[NT];
    // A 16x32 bf16: lane(m,lh): elems0-7 = K[+8*lh..], elems8-15 = K[+16+8*lh..]
#pragma unroll
    for (int mt = 0; mt < MT; ++mt) {
      ah[mt].q[0] = *(const uint4*)(pAH + mt * 16 * KC);
      ah[mt].q[1] = *(const uint4*)(pAH + mt * 16 * KC + 16);
      al[mt].q[0] = *(const uint4*)(pAL + mt * 16 * KC);
      al[mt].q[1] = *(const uint4*)(pAL + mt * 16 * KC + 16);
    }
    // B 32x16 bf16: lane(n,lh): elems0-7 = K[+8*lh..], elems8-15 = K[+16+8*lh..] of column n (= WT row n)
#pragma unroll
    for (int nt = 0; nt < NT; ++nt) {
      bh[nt].q[0] = *(const uint4*)(pBH + nt * 16 * KC);
      bh[nt].q[1] = *(const uint4*)(pBH + nt * 16 * KC + 16);
      bl[nt].q[0] = *(const uint4*)(pBL + nt * 16 * KC);
      bl[nt].q[1] = *(const uint4*)(pBL + nt * 16 * KC + 16);
    }
#pragma unroll
    for (int mt = 0; mt < MT; ++mt)
#pragma unroll
      for (int nt = 0; nt < NT; ++nt) {
        acc[mt][nt] = __builtin_amdgcn_wmma_f32_16x16x32_bf16(
            false, ah[mt].v, false, bh[nt].v, (short)0, acc[mt][nt], false, false);
        acc[mt][nt] = __builtin_amdgcn_wmma_f32_16x16x32_bf16(
            false, ah[mt].v, false, bl[nt].v, (short)0, acc[mt][nt], false, false);
        acc[mt][nt] = __builtin_amdgcn_wmma_f32_16x16x32_bf16(
            false, al[mt].v, false, bh[nt].v, (short)0, acc[mt][nt], false, false);
      }
    pAH += 32; pAL += 32; pBH += 32; pBL += 32;
  }

#pragma unroll
  for (int nt = 0; nt < NT; ++nt) {
    int n = n0 + nt * 16 + lm;
    float bn = bias[n];
#pragma unroll
    for (int mt = 0; mt < MT; ++mt) {
#pragma unroll
      for (int r = 0; r < 8; ++r) {
        // C layout: lanes 0-15 -> M=r, lanes 16-31 -> M=8+r (N = lane&15)
        int m = m0 + mt * 16 + r + 8 * lh;
        float v = fmaxf(acc[mt][nt][r] + bn, 0.0f);
        if (MODE == 0) {
          unsigned short hi, lo;
          split_bf16(v, hi, lo);
          size_t o = (size_t)m * NCOL + n;
          out_hi[o] = hi; out_lo[o] = lo;
        } else {
          out_f[(size_t)m * NCOL + n] = v;
        }
      }
    }
  }
}

// =============== 4. heads: cls/bbox dots + softmax + decode ===============
__global__ void head_kernel(const float* __restrict__ h2,
                            const float* __restrict__ cls_w,
                            const float* __restrict__ cls_b,
                            const float* __restrict__ bbox_w,
                            const float* __restrict__ bbox_b,
                            const float* __restrict__ boxes,
                            float* __restrict__ out) {
  __shared__ float red[10][128];
  int n = blockIdx.x;
  int tid = threadIdx.x;
  float acc[10];
#pragma unroll
  for (int j = 0; j < 10; ++j) acc[j] = 0.f;
  const float* hrow = h2 + (size_t)n * HID;
  for (int k = tid; k < HID; k += 128) {
    float hv = hrow[k];
    acc[0] += hv * cls_w[k * 2 + 0];
    acc[1] += hv * cls_w[k * 2 + 1];
#pragma unroll
    for (int j = 0; j < 8; ++j) acc[2 + j] += hv * bbox_w[k * 8 + j];
  }
#pragma unroll
  for (int j = 0; j < 10; ++j) red[j][tid] = acc[j];
  __syncthreads();
  if (tid < 10) {
    float s = 0.f;
    for (int t = 0; t < 128; ++t) s += red[tid][t];
    red[tid][0] = s;
  }
  __syncthreads();
  if (tid == 0) {
    float l0 = red[0][0] + cls_b[0];
    float l1 = red[1][0] + cls_b[1];
    float mx = fmaxf(l0, l1);
    float e0 = __expf(l0 - mx), e1 = __expf(l1 - mx);
    float inv = 1.f / (e0 + e1);
    float d[8];
#pragma unroll
    for (int j = 0; j < 8; ++j) d[j] = red[2 + j][0] + bbox_b[j];
    float bx1 = boxes[n * 4 + 0], by1 = boxes[n * 4 + 1];
    float bx2 = boxes[n * 4 + 2], by2 = boxes[n * 4 + 3];
    float pw = bx2 - bx1, ph = by2 - by1;
    float cx = bx1 + 0.5f * pw, cy = by1 + 0.5f * ph;
    const float CLAMPV = 4.135166556742356f;   // log(1000/16)
#pragma unroll
    for (int kc = 0; kc < 2; ++kc) {
      float dx = d[kc * 4 + 0] * 0.1f, dy = d[kc * 4 + 1] * 0.1f;
      float dw = fminf(d[kc * 4 + 2] * 0.2f, CLAMPV);
      float dh = fminf(d[kc * 4 + 3] * 0.2f, CLAMPV);
      float pcx = dx * pw + cx, pcy = dy * ph + cy;
      float pww = __expf(dw) * pw, phh = __expf(dh) * ph;
      out[n * 10 + kc * 4 + 0] = fminf(fmaxf(pcx - 0.5f * pww, 0.f), IMGW);
      out[n * 10 + kc * 4 + 1] = fminf(fmaxf(pcy - 0.5f * phh, 0.f), IMGH);
      out[n * 10 + kc * 4 + 2] = fminf(fmaxf(pcx + 0.5f * pww, 0.f), IMGW);
      out[n * 10 + kc * 4 + 3] = fminf(fmaxf(pcy + 0.5f * phh, 0.f), IMGH);
    }
    out[n * 10 + 8] = e0 * inv;
    out[n * 10 + 9] = e1 * inv;
  }
}

// ============================ launch driver ============================
extern "C" void kernel_launch(void* const* d_in, const int* in_sizes, int n_in,
                              void* d_out, int out_size, void* d_ws, size_t ws_size,
                              hipStream_t stream) {
  (void)in_sizes; (void)n_in; (void)out_size; (void)ws_size;
  const float* features = (const float*)d_in[0];
  const float* boxes    = (const float*)d_in[1];
  const int*   roib     = (const int*)d_in[2];
  const float* fc1_w    = (const float*)d_in[3];
  const float* fc1_b    = (const float*)d_in[4];
  const float* fc2_w    = (const float*)d_in[5];
  const float* fc2_b    = (const float*)d_in[6];
  const float* cls_w    = (const float*)d_in[7];
  const float* cls_b    = (const float*)d_in[8];
  const float* bbox_w   = (const float*)d_in[9];
  const float* bbox_b   = (const float*)d_in[10];
  float* out = (float*)d_out;

  char* ws = (char*)d_ws;
  size_t off = 0;
  auto wsalloc = [&](size_t bytes) -> void* {
    void* p = (void*)(ws + off);
    off += (bytes + 255) & ~(size_t)255;
    return p;
  };
  // ~114 MB total scratch
  unsigned short* wt1_hi = (unsigned short*)wsalloc((size_t)HID * D_FEAT * 2);
  unsigned short* wt1_lo = (unsigned short*)wsalloc((size_t)HID * D_FEAT * 2);
  unsigned short* wt2_hi = (unsigned short*)wsalloc((size_t)HID * HID * 2);
  unsigned short* wt2_lo = (unsigned short*)wsalloc((size_t)HID * HID * 2);
  unsigned short* f_hi   = (unsigned short*)wsalloc((size_t)NPAD * D_FEAT * 2);
  unsigned short* f_lo   = (unsigned short*)wsalloc((size_t)NPAD * D_FEAT * 2);
  unsigned short* h1_hi  = (unsigned short*)wsalloc((size_t)NPAD * HID * 2);
  unsigned short* h1_lo  = (unsigned short*)wsalloc((size_t)NPAD * HID * 2);
  float*          h2     = (float*)wsalloc((size_t)NPAD * HID * 4);

  dim3 tb(32, 8);
  transpose_split_kernel<<<dim3(D_FEAT / 32, HID / 32), tb, 0, stream>>>(
      fc1_w, wt1_hi, wt1_lo, D_FEAT, HID);
  transpose_split_kernel<<<dim3(HID / 32, HID / 32), tb, 0, stream>>>(
      fc2_w, wt2_hi, wt2_lo, HID, HID);

  roi_align_kernel<<<NPAD, 256, 0, stream>>>(features, boxes, roib, f_hi, f_lo);

  gemm_wmma_kernel<0, D_FEAT>
      <<<dim3(NPAD / (16 * MT), HID / (16 * NT)), 32, 0, stream>>>(
          f_hi, f_lo, wt1_hi, wt1_lo, fc1_b, h1_hi, h1_lo, nullptr, HID);
  gemm_wmma_kernel<1, HID>
      <<<dim3(NPAD / (16 * MT), HID / (16 * NT)), 32, 0, stream>>>(
          h1_hi, h1_lo, wt2_hi, wt2_lo, fc2_b, nullptr, nullptr, h2, HID);

  head_kernel<<<ROI_N, 128, 0, stream>>>(h2, cls_w, cls_b, bbox_w, bbox_b,
                                         boxes, out);
}
